// HepatotoxicityGAT_30279519437232
// MI455X (gfx1250) — matrix operand order repaired and checked by reading, not compiled, not run on hardware
//
#include <hip/hip_runtime.h>

// ---------------------------------------------------------------------------
// Types for CDNA5 WMMA (wave32, 16x16x32 bf16 -> f32)
// ---------------------------------------------------------------------------
typedef __attribute__((ext_vector_type(16))) __bf16 v16bf;
typedef __attribute__((ext_vector_type(8)))  __bf16 v8bf;
typedef __attribute__((ext_vector_type(8)))  float  v8f;

// float -> bf16 (round to nearest even), raw u16 storage (no __bf16 arith).
__device__ __forceinline__ unsigned short f2bf(float f) {
    unsigned int u = __float_as_uint(f);
    unsigned int r = u + 0x7FFFu + ((u >> 16) & 1u);
    return (unsigned short)(r >> 16);
}

// order-preserving float<->uint mapping for atomicMax-based segment max
__device__ __forceinline__ unsigned int f2ord(float f) {
    unsigned int u = __float_as_uint(f);
    return (u >> 31) ? ~u : (u | 0x80000000u);
}
__device__ __forceinline__ float ord2f(unsigned int u) {
    unsigned int v = (u >> 31) ? (u & 0x7FFFFFFFu) : ~u;
    return __uint_as_float(v);
}

// ---------------------------------------------------------------------------
// Utility kernels
// ---------------------------------------------------------------------------
__global__ void gat_fill_u32(unsigned int* p, unsigned int v, size_t n) {
    size_t i = (size_t)blockIdx.x * blockDim.x + threadIdx.x;
    if (i < n) p[i] = v;
}

__global__ void gat_to_bf16(const float* __restrict__ src,
                            unsigned short* __restrict__ dst, size_t n) {
    size_t i = (size_t)blockIdx.x * blockDim.x + threadIdx.x;
    if (i < n) dst[i] = f2bf(src[i]);
}

// W [K, Nc] f32 row-major  ->  WT [Nc, K] bf16 (transposed, contiguous in K)
__global__ void gat_w_transpose_bf16(const float* __restrict__ W,
                                     unsigned short* __restrict__ WT,
                                     int K, int Nc) {
    int idx = blockIdx.x * blockDim.x + threadIdx.x;
    if (idx >= K * Nc) return;
    int k = idx / Nc, n = idx % Nc;
    WT[(size_t)n * K + k] = f2bf(W[idx]);
}

// ---------------------------------------------------------------------------
// BF16 WMMA GEMM: C[M,Nc] = A[M,K] * B[K,Nc]; A bf16 row-major, B as BT[Nc,K].
// One wave computes a 16x64 strip: 4 accumulators sharing one A fragment per
// K-step (A-fragment register reuse). A/B use contiguous k order (k = 16*half + s).
// Requires Nc % 64 == 0 (true here: 256 and 64) and M % 16 == 0.
// ---------------------------------------------------------------------------
__global__ __launch_bounds__(256)
void gat_gemm_bf16_wmma(const unsigned short* __restrict__ A,
                        const unsigned short* __restrict__ BT,
                        float* __restrict__ C,
                        int M, int K, int Nc) {
    const int wave    = (int)((blockIdx.x * blockDim.x + threadIdx.x) >> 5);
    const int lane    = threadIdx.x & 31;
    const int ngroups = Nc >> 6;               // groups of 4 N-tiles
    const int totalw  = (M >> 4) * ngroups;
    if (wave >= totalw) return;                // uniform per-wave exit
    const int tm  = wave / ngroups;
    const int tn0 = (wave % ngroups) << 2;     // first of 4 N tiles

    const int half = lane >> 4;                // 0: lanes 0-15, 1: lanes 16-31
    const int lid  = lane & 15;

    const unsigned short* arow = A  + (size_t)(tm * 16 + lid) * K;
    const unsigned short* brow = BT + (size_t)(tn0 * 16 + lid) * K;
    const size_t bstride = (size_t)16 * K;     // BT rows between adjacent tiles

    v8f acc0 = {0.f,0.f,0.f,0.f,0.f,0.f,0.f,0.f};
    v8f acc1 = acc0, acc2 = acc0, acc3 = acc0;

    for (int kb = 0; kb < K; kb += 32) {
        // A frag: elems 0..15 = K kb+16*half .. +15 (contiguous per row)
        v8bf a_lo = *(const v8bf*)(arow + kb + half * 16);
        v8bf a_hi = *(const v8bf*)(arow + kb + half * 16 + 8);
        v16bf av  = __builtin_shufflevector(a_lo, a_hi,
            0,1,2,3,4,5,6,7,8,9,10,11,12,13,14,15);

        // 4 B frags: elems 0..15 = K kb+16*half .. +15 (contiguous per row)
        const unsigned short* bk = brow + kb + half * 16;
        v8bf b0l = *(const v8bf*)(bk);
        v8bf b0h = *(const v8bf*)(bk + 8);
        v8bf b1l = *(const v8bf*)(bk + bstride);
        v8bf b1h = *(const v8bf*)(bk + bstride + 8);
        v8bf b2l = *(const v8bf*)(bk + 2 * bstride);
        v8bf b2h = *(const v8bf*)(bk + 2 * bstride + 8);
        v8bf b3l = *(const v8bf*)(bk + 3 * bstride);
        v8bf b3h = *(const v8bf*)(bk + 3 * bstride + 8);

        v16bf bv0 = __builtin_shufflevector(b0l, b0h,
            0,1,2,3,4,5,6,7,8,9,10,11,12,13,14,15);
        v16bf bv1 = __builtin_shufflevector(b1l, b1h,
            0,1,2,3,4,5,6,7,8,9,10,11,12,13,14,15);
        v16bf bv2 = __builtin_shufflevector(b2l, b2h,
            0,1,2,3,4,5,6,7,8,9,10,11,12,13,14,15);
        v16bf bv3 = __builtin_shufflevector(b3l, b3h,
            0,1,2,3,4,5,6,7,8,9,10,11,12,13,14,15);

        acc0 = __builtin_amdgcn_wmma_f32_16x16x32_bf16(
            false, av, false, bv0, (short)0, acc0, false, false);
        acc1 = __builtin_amdgcn_wmma_f32_16x16x32_bf16(
            false, av, false, bv1, (short)0, acc1, false, false);
        acc2 = __builtin_amdgcn_wmma_f32_16x16x32_bf16(
            false, av, false, bv2, (short)0, acc2, false, false);
        acc3 = __builtin_amdgcn_wmma_f32_16x16x32_bf16(
            false, av, false, bv3, (short)0, acc3, false, false);
    }

    // C/D layout: VGPR r holds M = r + 8*half, N = lid (+ 16 per N tile)
    float* cbase = C + (size_t)(tm * 16 + half * 8) * Nc + tn0 * 16 + lid;
#pragma unroll
    for (int r = 0; r < 8; ++r) {
        float* cr = cbase + (size_t)r * Nc;
        cr[0]  = acc0[r];
        cr[16] = acc1[r];
        cr[32] = acc2[r];
        cr[48] = acc3[r];
    }
}

// ---------------------------------------------------------------------------
// asrc[n,h] = sum_c g[n,h*C+c]*as[h,c];  adst likewise. C fixed at 64.
// ---------------------------------------------------------------------------
__global__ void gat_attn_coeff(const float* __restrict__ g,
                               const float* __restrict__ as,
                               const float* __restrict__ ad,
                               float* __restrict__ asrc,
                               float* __restrict__ adst,
                               int Nn, int Hh) {
    int idx = blockIdx.x * blockDim.x + threadIdx.x;
    if (idx >= Nn * Hh) return;
    int n = idx / Hh, h = idx % Hh;
    const float* row = g + (size_t)n * Hh * 64 + h * 64;
    const float* sa  = as + h * 64;
    const float* da  = ad + h * 64;
    float s = 0.f, d = 0.f;
#pragma unroll 8
    for (int c = 0; c < 64; ++c) { float v = row[c]; s += v * sa[c]; d += v * da[c]; }
    asrc[idx] = s;
    adst[idx] = d;
}

// ---------------------------------------------------------------------------
// Edge pass 1: logits + segment-max via ordered-uint atomicMax
// ---------------------------------------------------------------------------
__global__ void gat_edge_logits(const long long* __restrict__ ei,
                                const float* __restrict__ asrc,
                                const float* __restrict__ adst,
                                float* __restrict__ elog,
                                unsigned int* __restrict__ mmax,
                                int Ee, int Nn, int Hh) {
    int e = blockIdx.x * blockDim.x + threadIdx.x;
    int Ep = Ee + Nn;
    if (e >= Ep) return;
    int s, d;
    if (e < Ee) { s = (int)ei[e]; d = (int)ei[(size_t)Ee + e]; }
    else        { s = d = e - Ee; }
    for (int h = 0; h < Hh; ++h) {
        float x = asrc[(size_t)s * Hh + h] + adst[(size_t)d * Hh + h];
        x = (x > 0.f) ? x : 0.2f * x;                    // leaky_relu(0.2)
        elog[(size_t)e * Hh + h] = x;
        atomicMax(&mmax[(size_t)d * Hh + h], f2ord(x));
    }
}

// ---------------------------------------------------------------------------
// Edge pass 2: e = exp(logit - max[dst]); denom[dst] += e
// ---------------------------------------------------------------------------
__global__ void gat_edge_exp(const long long* __restrict__ ei,
                             float* __restrict__ elog,
                             const unsigned int* __restrict__ mmax,
                             float* __restrict__ denom,
                             int Ee, int Nn, int Hh) {
    int e = blockIdx.x * blockDim.x + threadIdx.x;
    int Ep = Ee + Nn;
    if (e >= Ep) return;
    int d;
    if (e < Ee) d = (int)ei[(size_t)Ee + e];
    else        d = e - Ee;
    for (int h = 0; h < Hh; ++h) {
        float m = ord2f(mmax[(size_t)d * Hh + h]);
        float v = __expf(elog[(size_t)e * Hh + h] - m);
        elog[(size_t)e * Hh + h] = v;
        atomicAdd(&denom[(size_t)d * Hh + h], v);
    }
}

// ---------------------------------------------------------------------------
// Edge pass 3: out[dst] += alpha * g[src]  (4 channels / thread, float4 gather)
// ---------------------------------------------------------------------------
__global__ void gat_edge_msg(const long long* __restrict__ ei,
                             const float* __restrict__ g,
                             const float* __restrict__ elog,
                             const float* __restrict__ denom,
                             float* __restrict__ out,
                             int Ee, int Nn, int Hh, int HC) {
    const int chunks = HC >> 2;
    size_t idx = (size_t)blockIdx.x * blockDim.x + threadIdx.x;
    size_t total = (size_t)(Ee + Nn) * chunks;
    if (idx >= total) return;
    int e  = (int)(idx / chunks);
    int c  = (int)(idx % chunks) * 4;
    int h  = c >> 6;                                    // C = 64 per head
    int s, d;
    if (e < Ee) { s = (int)ei[e]; d = (int)ei[(size_t)Ee + e]; }
    else        { s = d = e - Ee; }
    float alpha = elog[(size_t)e * Hh + h] /
                  (denom[(size_t)d * Hh + h] + 1e-16f);
    const float4 gv = *(const float4*)(g + (size_t)s * HC + c);
    float* ob = out + (size_t)d * HC + c;
    atomicAdd(ob + 0, gv.x * alpha);
    atomicAdd(ob + 1, gv.y * alpha);
    atomicAdd(ob + 2, gv.z * alpha);
    atomicAdd(ob + 3, gv.w * alpha);
}

// ---------------------------------------------------------------------------
// bias add + optional ELU; also emits the bf16 copy for the next layer's GEMM
// (fused to avoid a separate full-tensor conversion pass)
// ---------------------------------------------------------------------------
__global__ void gat_bias_act(float* __restrict__ h, const float* __restrict__ b,
                             unsigned short* __restrict__ bfout,
                             size_t total, int HC, int do_elu) {
    size_t idx = (size_t)blockIdx.x * blockDim.x + threadIdx.x;
    if (idx >= total) return;
    int c = (int)(idx % HC);
    float v = h[idx] + b[c];
    if (do_elu) v = (v > 0.f) ? v : (__expf(v) - 1.f);
    h[idx] = v;
    if (bfout) bfout[idx] = f2bf(v);
}

// ---------------------------------------------------------------------------
// global mean pool: accumulate sums [G,64] and counts [G]
// ---------------------------------------------------------------------------
__global__ void gat_pool(const float* __restrict__ h3,
                         const long long* __restrict__ batch,
                         float* __restrict__ sums, float* __restrict__ cnt,
                         int Nn) {
    size_t idx = (size_t)blockIdx.x * blockDim.x + threadIdx.x;
    if (idx >= (size_t)Nn * 64) return;
    int n = (int)(idx >> 6);
    int c = (int)(idx & 63);
    int gI = (int)batch[n];
    atomicAdd(&sums[(size_t)gI * 64 + c], h3[idx]);
    if (c == 0) atomicAdd(&cnt[gI], 1.f);
}

// ---------------------------------------------------------------------------
// classifier: pooled(64) -> relu(32) -> 1   (one thread per graph)
// ---------------------------------------------------------------------------
__global__ void gat_classifier(const float* __restrict__ sums,
                               const float* __restrict__ cnt,
                               const float* __restrict__ Wc1,
                               const float* __restrict__ bc1,
                               const float* __restrict__ Wc2,
                               const float* __restrict__ bc2,
                               float* __restrict__ out, int Gn) {
    int gI = blockIdx.x * blockDim.x + threadIdx.x;
    if (gI >= Gn) return;
    float inv = 1.f / fmaxf(cnt[gI], 1.f);
    float p[64];
#pragma unroll
    for (int c = 0; c < 64; ++c) p[c] = sums[(size_t)gI * 64 + c] * inv;
    float o = bc2[0];
    for (int j = 0; j < 32; ++j) {
        float a = bc1[j];
#pragma unroll 8
        for (int c = 0; c < 64; ++c) a += p[c] * Wc1[c * 32 + j];
        a = fmaxf(a, 0.f);
        o += a * Wc2[j];
    }
    out[gI] = o;
}

// ---------------------------------------------------------------------------
// Host orchestration
// ---------------------------------------------------------------------------
static inline unsigned int nblk(size_t total, int tpb) {
    return (unsigned int)((total + tpb - 1) / tpb);
}

extern "C" void kernel_launch(void* const* d_in, const int* in_sizes, int n_in,
                              void* d_out, int out_size, void* d_ws, size_t ws_size,
                              hipStream_t stream) {
    const float*     x     = (const float*)d_in[0];
    const long long* ei    = (const long long*)d_in[1];
    const long long* batch = (const long long*)d_in[2];
    const float* W1  = (const float*)d_in[3];
    const float* a1s = (const float*)d_in[4];
    const float* a1d = (const float*)d_in[5];
    const float* b1  = (const float*)d_in[6];
    const float* W2  = (const float*)d_in[7];
    const float* a2s = (const float*)d_in[8];
    const float* a2d = (const float*)d_in[9];
    const float* b2  = (const float*)d_in[10];
    const float* W3  = (const float*)d_in[11];
    const float* a3s = (const float*)d_in[12];
    const float* a3d = (const float*)d_in[13];
    const float* b3  = (const float*)d_in[14];
    const float* Wc1 = (const float*)d_in[15];
    const float* bc1 = (const float*)d_in[16];
    const float* Wc2 = (const float*)d_in[17];
    const float* bc2 = (const float*)d_in[18];

    const int Nn = in_sizes[2];          // batch has N elements
    const int Ee = in_sizes[1] / 2;      // edge_index is [2,E]
    const int Ep = Ee + Nn;              // with self loops
    const int Gn = out_size;             // [G,1] output

    // ---- carve workspace ----
    char* ws = (char*)d_ws;
    auto carve = [&](size_t bytes) -> char* {
        char* p = ws;
        ws += (bytes + 255) & ~(size_t)255;
        return p;
    };
    float*          buf0  = (float*)carve((size_t)Nn * 256 * 4);  // h / out acc
    float*          buf1  = (float*)carve((size_t)Nn * 256 * 4);  // g
    unsigned short* bfbuf = (unsigned short*)carve((size_t)Nn * 256 * 2);
    unsigned short* WT    = (unsigned short*)carve((size_t)256 * 256 * 2);
    float*          asrcb = (float*)carve((size_t)Nn * 4 * 4);
    float*          adstb = (float*)carve((size_t)Nn * 4 * 4);
    unsigned int*   mmax  = (unsigned int*)carve((size_t)Nn * 4 * 4);
    float*          denom = (float*)carve((size_t)Nn * 4 * 4);
    float*          elog  = (float*)carve((size_t)Ep * 4 * 4);
    float*          sums  = (float*)carve((size_t)Gn * 64 * 4);
    float*          cnt   = (float*)carve((size_t)Gn * 4);
    (void)ws_size;

    const int TPB = 256;

    // have_bf16: bfbuf already holds the bf16 copy of the layer input
    // (written by the previous layer's fused bias/ELU epilogue)
    auto layer = [&](const float* hin, int have_bf16, int Kin, int HC, int Hh,
                     const float* W, const float* as_, const float* ad_,
                     const float* bias, int do_elu,
                     unsigned short* bf_next) {
        // 1. activations -> bf16 (only when not produced by previous epilogue)
        if (!have_bf16) {
            size_t nK = (size_t)Nn * Kin;
            gat_to_bf16<<<nblk(nK, TPB), TPB, 0, stream>>>(hin, bfbuf, nK);
        }
        // 2. weights -> transposed bf16
        gat_w_transpose_bf16<<<nblk((size_t)Kin * HC, TPB), TPB, 0, stream>>>(
            W, WT, Kin, HC);
        // 3. g = h @ W via WMMA: one wave per 16x64 strip (4 tiles, shared A)
        int waves = (Nn / 16) * (HC / 64);
        gat_gemm_bf16_wmma<<<(unsigned)((waves + 7) / 8), TPB, 0, stream>>>(
            bfbuf, WT, buf1, Nn, Kin, HC);
        // 4. attention projections
        gat_attn_coeff<<<nblk((size_t)Nn * Hh, TPB), TPB, 0, stream>>>(
            buf1, as_, ad_, asrcb, adstb, Nn, Hh);
        // 5. zero segment-max (ordered -inf == 0u), denom, out accumulator
        gat_fill_u32<<<nblk((size_t)Nn * Hh, TPB), TPB, 0, stream>>>(
            mmax, 0u, (size_t)Nn * Hh);
        gat_fill_u32<<<nblk((size_t)Nn * Hh, TPB), TPB, 0, stream>>>(
            (unsigned int*)denom, 0u, (size_t)Nn * Hh);
        gat_fill_u32<<<nblk((size_t)Nn * HC, TPB), TPB, 0, stream>>>(
            (unsigned int*)buf0, 0u, (size_t)Nn * HC);
        // 6. segment softmax + message scatter
        gat_edge_logits<<<nblk((size_t)Ep, TPB), TPB, 0, stream>>>(
            ei, asrcb, adstb, elog, mmax, Ee, Nn, Hh);
        gat_edge_exp<<<nblk((size_t)Ep, TPB), TPB, 0, stream>>>(
            ei, elog, mmax, denom, Ee, Nn, Hh);
        gat_edge_msg<<<nblk((size_t)Ep * (HC / 4), TPB), TPB, 0, stream>>>(
            ei, buf1, elog, denom, buf0, Ee, Nn, Hh, HC);
        // 7. bias + activation (+ fused bf16 emit for next layer's GEMM)
        gat_bias_act<<<nblk((size_t)Nn * HC, TPB), TPB, 0, stream>>>(
            buf0, bias, bf_next, (size_t)Nn * HC, HC, do_elu);
    };

    layer(x,    0, 64,  256, 4, W1, a1s, a1d, b1, 1, bfbuf);
    layer(buf0, 1, 256, 256, 4, W2, a2s, a2d, b2, 1, bfbuf);
    layer(buf0, 1, 256, 64,  1, W3, a3s, a3d, b3, 0, nullptr);

    // global mean pool + classifier
    gat_fill_u32<<<nblk((size_t)Gn * 64, TPB), TPB, 0, stream>>>(
        (unsigned int*)sums, 0u, (size_t)Gn * 64);
    gat_fill_u32<<<nblk((size_t)Gn, TPB), TPB, 0, stream>>>(
        (unsigned int*)cnt, 0u, (size_t)Gn);
    gat_pool<<<nblk((size_t)Nn * 64, TPB), TPB, 0, stream>>>(
        buf0, batch, sums, cnt, Nn);
    gat_classifier<<<nblk((size_t)Gn, TPB), TPB, 0, stream>>>(
        sums, cnt, Wc1, bc1, Wc2, bc2, (float*)d_out, Gn);
}
